// S5Backbone_38328288149699
// MI455X (gfx1250) — matrix operand order repaired and checked by reading, not compiled, not run on hardware
//
#include <hip/hip_runtime.h>
#include <hip/hip_bf16.h>
#include <math.h>
#include <stdint.h>

// Problem dims (fixed by the reference)
#define N_B     16384
#define N_DETER 2048
#define N_STOCH 1024
#define N_ACT   12
#define N_HID   1024
#define N_RANK  16
#define K1_PAD  1056          // 1036 padded up to a multiple of 32
#define EPSV    1e-4f

typedef __attribute__((ext_vector_type(16))) __bf16          bf16x16;
typedef __attribute__((ext_vector_type(8)))  float           f32x8;
typedef __attribute__((ext_vector_type(8)))  unsigned short  u16x8;

union FragBF { bf16x16 v; u16x8 h[2]; };

__device__ __forceinline__ unsigned short f2bf(float f) {
  union { float f; unsigned u; } c; c.f = f;
  return (unsigned short)((c.u + 0x7FFFu + ((c.u >> 16) & 1u)) >> 16); // RNE
}
__device__ __forceinline__ float softplus_f(float x) {
  return (x > 20.f) ? x : log1pf(expf(x));
}
__device__ __forceinline__ float sigmoid_f(float x) { return 1.f / (1.f + expf(-x)); }

// Async global->LDS 16B copy (CDNA5 GLOBAL_LOAD_ASYNC_TO_LDS_B128, ASYNCcnt).
__device__ __forceinline__ void async_cp16(unsigned lds_byte_addr, const void* gaddr) {
  asm volatile("global_load_async_to_lds_b128 %0, %1, off"
               :: "v"(lds_byte_addr), "v"(gaddr) : "memory");
}
__device__ __forceinline__ unsigned lds_addr_of(const void* p) {
  return (unsigned)(unsigned long long)p; // low 32 bits of generic ptr = LDS offset
}

// ---------------------------------------------------------------------------
// bf16 WMMA GEMM: C[M,N] = A[M,K](bf16, row-major) * Bt[N,K](bf16, N-major) + bias[N]
// Block tile 128x256, BK=32. 256 threads = 8 waves (2x4); wave tile 64x64 (4x4 WMMA).
// Double-buffered LDS fed by async global->LDS copies.
// M mult of 128, N mult of 256, K mult of 32.
// ---------------------------------------------------------------------------
__launch_bounds__(256)
__global__ void gemm_bf16_wmma(const unsigned short* __restrict__ A,
                               const unsigned short* __restrict__ Bt,
                               const float* __restrict__ bias,
                               float* __restrict__ C,
                               int M, int N, int K)
{
  __shared__ __align__(16) unsigned short lA[2][128 * 32]; // [m][k]
  __shared__ __align__(16) unsigned short lB[2][256 * 32]; // [n][k]

  const int tid   = threadIdx.x;
  const int wave  = tid >> 5;
  const int lane  = tid & 31;
  const int wm0   = (wave >> 2) * 64;   // 2 waves along M
  const int wn0   = (wave & 3) * 64;    // 4 waves along N
  const int m0    = blockIdx.x * 128;
  const int n0    = blockIdx.y * 256;
  const int half  = lane >> 4;          // 0: lanes 0-15, 1: lanes 16-31
  const int lan16 = lane & 15;

  f32x8 acc[4][4];
  const f32x8 zero8 = {0.f,0.f,0.f,0.f,0.f,0.f,0.f,0.f};
  #pragma unroll
  for (int i = 0; i < 4; ++i)
    #pragma unroll
    for (int j = 0; j < 4; ++j) acc[i][j] = zero8;

  const unsigned short* gA0 = A  + (size_t)m0 * K;
  const unsigned short* gB0 = Bt + (size_t)n0 * K;

  // Issue async stage of one K-tile into buffer `buf` (6 x b128 per thread).
  auto stage = [&](int buf, int k0) {
    #pragma unroll
    for (int cc = 0; cc < 2; ++cc) {               // A: 128 rows x 32 k = 512 chunks
      int c = tid + cc * 256;
      int row = c >> 2, koff = (c & 3) * 8;
      async_cp16(lds_addr_of(&lA[buf][row * 32 + koff]),
                 gA0 + (size_t)row * K + k0 + koff);
    }
    #pragma unroll
    for (int cc = 0; cc < 4; ++cc) {               // B: 256 rows x 32 k = 1024 chunks
      int c = tid + cc * 256;
      int n = c >> 2, koff = (c & 3) * 8;
      async_cp16(lds_addr_of(&lB[buf][n * 32 + koff]),
                 gB0 + (size_t)n * K + k0 + koff);
    }
  };

  const int nstages = K >> 5;
  stage(0, 0);
  int buf = 0;
  for (int s = 0; s < nstages; ++s) {
    if (s + 1 < nstages) {
      stage(buf ^ 1, (s + 1) << 5);
      asm volatile("s_wait_asynccnt 0x6" ::: "memory"); // drain the 6 older copies
    } else {
      asm volatile("s_wait_asynccnt 0x0" ::: "memory");
    }
    __syncthreads();   // all waves' tiles visible

    // B fragments: lane holds column n; K contiguous per half-wave (same k order as A).
    FragBF fb[4];
    #pragma unroll
    for (int nt = 0; nt < 4; ++nt) {
      int n = wn0 + nt * 16 + lan16;
      fb[nt].h[0] = *(const u16x8*)(&lB[buf][n * 32 + half * 16]);
      fb[nt].h[1] = *(const u16x8*)(&lB[buf][n * 32 + half * 16 + 8]);
    }
    // A fragments: lane holds row m; K contiguous per half-wave (0..15 half 0, 16..31 half 1).
    #pragma unroll
    for (int mt = 0; mt < 4; ++mt) {
      FragBF fa;
      int m = wm0 + mt * 16 + lan16;
      fa.h[0] = *(const u16x8*)(&lA[buf][m * 32 + half * 16]);
      fa.h[1] = *(const u16x8*)(&lA[buf][m * 32 + half * 16 + 8]);
      #pragma unroll
      for (int nt = 0; nt < 4; ++nt) {
        acc[mt][nt] = __builtin_amdgcn_wmma_f32_16x16x32_bf16(
            false, fa.v, false, fb[nt].v, (short)0, acc[mt][nt], false, false);
      }
    }
    __syncthreads();   // done reading `buf`; next iter may overwrite it
    buf ^= 1;
  }

  // Epilogue: C/D layout — VGPR r holds M = r + half*8; N = lane%16.
  #pragma unroll
  for (int mt = 0; mt < 4; ++mt) {
    #pragma unroll
    for (int nt = 0; nt < 4; ++nt) {
      int col = n0 + wn0 + nt * 16 + lan16;
      float bv = bias[col];
      int rowbase = m0 + wm0 + mt * 16 + half * 8;
      #pragma unroll
      for (int r = 0; r < 8; ++r) {
        C[(size_t)(rowbase + r) * N + col] = acc[mt][nt][r] + bv;
      }
    }
  }
}

// ---------------------------------------------------------------------------
// Prep: per-channel dt = softplus(dt_p)+1e-4 ; decay = exp(-softplus(a_log)*dt)
// ---------------------------------------------------------------------------
__launch_bounds__(256)
__global__ void prep_vec_kernel(const float* __restrict__ a_log,
                                const float* __restrict__ dt_p,
                                float* __restrict__ dtv, float* __restrict__ dcay)
{
  int i = blockIdx.x * 256 + threadIdx.x;
  if (i >= N_DETER) return;
  float dt = softplus_f(dt_p[i]) + 1e-4f;
  float a  = -softplus_f(a_log[i]);
  dtv[i]  = dt;
  dcay[i] = expf(a * dt);
}

// Pack [stoch | action/max(|a|,1) | 0-pad] -> bf16 [B, K1_PAD]
__launch_bounds__(256)
__global__ void pack_x1_kernel(const float* __restrict__ stoch,
                               const float* __restrict__ action,
                               unsigned short* __restrict__ x1)
{
  size_t i = (size_t)blockIdx.x * 256 + threadIdx.x;
  if (i >= (size_t)N_B * K1_PAD) return;
  int row = (int)(i / K1_PAD);
  int col = (int)(i % K1_PAD);
  float v = 0.f;
  if (col < N_STOCH) {
    v = stoch[(size_t)row * N_STOCH + col];
  } else if (col < N_STOCH + N_ACT) {
    float a = action[(size_t)row * N_ACT + (col - N_STOCH)];
    v = a / fmaxf(fabsf(a), 1.f);
  }
  x1[i] = f2bf(v);
}

// Pack fp32 weight [Kin,N] -> TRANSPOSED bf16 [N,Kpad] (zero pad k)
__launch_bounds__(256)
__global__ void pack_wT_bf16(const float* __restrict__ w,
                             unsigned short* __restrict__ o,
                             int Kin, int Kpad, int N)
{
  size_t i = (size_t)blockIdx.x * 256 + threadIdx.x;
  if (i >= (size_t)N * Kpad) return;
  int n = (int)(i / Kpad);
  int k = (int)(i % Kpad);
  o[i] = (k < Kin) ? f2bf(w[(size_t)k * N + n]) : (unsigned short)0;
}

// RMSNorm(g1) + SiLU over rows of hpre [B,HID] -> bf16 activations
__launch_bounds__(256)
__global__ void act1_kernel(const float* __restrict__ h,
                            const float* __restrict__ g1,
                            unsigned short* __restrict__ x)
{
  __shared__ float red[256];
  int row = blockIdx.x;
  const float* hr = h + (size_t)row * N_HID;
  float ss = 0.f;
  for (int c = threadIdx.x; c < N_HID; c += 256) { float v = hr[c]; ss += v * v; }
  red[threadIdx.x] = ss;
  __syncthreads();
  for (int s = 128; s > 0; s >>= 1) {
    if (threadIdx.x < s) red[threadIdx.x] += red[threadIdx.x + s];
    __syncthreads();
  }
  float inv = rsqrtf(red[0] / (float)N_HID + EPSV);
  for (int c = threadIdx.x; c < N_HID; c += 256) {
    float v = hr[c] * inv * g1[c];
    x[(size_t)row * N_HID + c] = f2bf(v * sigmoid_f(v)); // SiLU
  }
}

// Low-rank mix + SSM update; writes joint = [ssm | inp] as bf16 [B, 2*DETER]
__launch_bounds__(256)
__global__ void lowrank_ssm_kernel(const float* __restrict__ deter,
                                   const float* __restrict__ Wu,
                                   const float* __restrict__ Wv,
                                   const float* __restrict__ dtv,
                                   const float* __restrict__ dcay,
                                   const float* __restrict__ inp,
                                   unsigned short* __restrict__ joint)
{
  __shared__ float ld[N_DETER];
  __shared__ float part[256];
  __shared__ float tj[N_RANK];
  int row = blockIdx.x;
  const float* dr = deter + (size_t)row * N_DETER;
  for (int c = threadIdx.x; c < N_DETER; c += 256) ld[c] = dr[c];
  __syncthreads();
  // t[j] = deter_row @ Wu[:,j] ; thread t: column j = t%16, K-slice t/16 (128 k's)
  int j  = threadIdx.x & 15;
  int sl = threadIdx.x >> 4;
  float p = 0.f;
  int kbeg = sl * 128;
  for (int k = kbeg; k < kbeg + 128; ++k) p += ld[k] * Wu[(size_t)k * N_RANK + j];
  part[threadIdx.x] = p;
  __syncthreads();
  if (threadIdx.x < N_RANK) {
    float s = 0.f;
    #pragma unroll
    for (int s2 = 0; s2 < 16; ++s2) s += part[s2 * 16 + threadIdx.x];
    tj[threadIdx.x] = s;
  }
  __syncthreads();
  const float* ir = inp + (size_t)row * N_DETER;
  unsigned short* jr = joint + (size_t)row * (2 * N_DETER);
  for (int c = threadIdx.x; c < N_DETER; c += 256) {
    float mix = ld[c];
    #pragma unroll
    for (int jj = 0; jj < N_RANK; ++jj) mix += tj[jj] * Wv[(size_t)jj * N_DETER + c];
    float iv  = ir[c];
    float ssm = dcay[c] * mix + dtv[c] * iv;
    jr[c]           = f2bf(ssm);
    jr[N_DETER + c] = f2bf(iv);
  }
}

// out = rmsnorm(sig(g)*tanh(c) + (1-sig(g))*deter, gn)
__launch_bounds__(256)
__global__ void final_kernel(const float* __restrict__ gatep,
                             const float* __restrict__ candp,
                             const float* __restrict__ deter,
                             const float* __restrict__ gn,
                             float* __restrict__ out)
{
  __shared__ float o[N_DETER];
  __shared__ float red[256];
  int row = blockIdx.x;
  const float* gp = gatep + (size_t)row * N_DETER;
  const float* cp = candp + (size_t)row * N_DETER;
  const float* dr = deter + (size_t)row * N_DETER;
  float ss = 0.f;
  for (int c = threadIdx.x; c < N_DETER; c += 256) {
    float g  = sigmoid_f(gp[c]);
    float cn = tanhf(cp[c]);
    float v  = g * cn + (1.f - g) * dr[c];
    o[c] = v;
    ss += v * v;
  }
  red[threadIdx.x] = ss;
  __syncthreads();
  for (int s = 128; s > 0; s >>= 1) {
    if (threadIdx.x < s) red[threadIdx.x] += red[threadIdx.x + s];
    __syncthreads();
  }
  float inv = rsqrtf(red[0] / (float)N_DETER + EPSV);
  for (int c = threadIdx.x; c < N_DETER; c += 256)
    out[(size_t)row * N_DETER + c] = o[c] * inv * gn[c];
}

// ---------------------------------------------------------------------------
extern "C" void kernel_launch(void* const* d_in, const int* in_sizes, int n_in,
                              void* d_out, int out_size, void* d_ws, size_t ws_size,
                              hipStream_t stream)
{
  (void)in_sizes; (void)n_in; (void)out_size; (void)ws_size;
  const float* stoch  = (const float*)d_in[0];
  const float* deter  = (const float*)d_in[1];
  const float* action = (const float*)d_in[2];
  const float* W1     = (const float*)d_in[3];
  const float* b1     = (const float*)d_in[4];
  const float* g1     = (const float*)d_in[5];
  const float* W2     = (const float*)d_in[6];
  const float* b2     = (const float*)d_in[7];
  const float* Wu     = (const float*)d_in[8];
  const float* Wv     = (const float*)d_in[9];
  const float* a_log  = (const float*)d_in[10];
  const float* dt_p   = (const float*)d_in[11];
  const float* Wg     = (const float*)d_in[12];
  const float* bg     = (const float*)d_in[13];
  const float* Wc     = (const float*)d_in[14];
  const float* bc     = (const float*)d_in[15];
  const float* gn     = (const float*)d_in[16];
  float* out = (float*)d_out;

  char* ws = (char*)d_ws;
  size_t off = 0;
  auto alloc = [&](size_t bytes) -> char* {
    char* p = ws + off;
    off += (bytes + 255) & ~(size_t)255;
    return p;
  };
  unsigned short* x1    = (unsigned short*)alloc((size_t)N_B * K1_PAD * 2);
  unsigned short* w1t   = (unsigned short*)alloc((size_t)N_HID * K1_PAD * 2);
  float*          hpre  = (float*)         alloc((size_t)N_B * N_HID * 4);
  unsigned short* xact  = (unsigned short*)alloc((size_t)N_B * N_HID * 2);
  unsigned short* w2t   = (unsigned short*)alloc((size_t)N_DETER * N_HID * 2);
  float*          inp   = (float*)         alloc((size_t)N_B * N_DETER * 4);
  unsigned short* joint = (unsigned short*)alloc((size_t)N_B * 2 * N_DETER * 2);
  unsigned short* wgt   = (unsigned short*)alloc((size_t)N_DETER * 2 * N_DETER * 2);
  unsigned short* wct   = (unsigned short*)alloc((size_t)N_DETER * 2 * N_DETER * 2);
  float*          gpre  = (float*)         alloc((size_t)N_B * N_DETER * 4);
  float*          cpre  = (float*)         alloc((size_t)N_B * N_DETER * 4);
  float*          dtv   = (float*)         alloc((size_t)N_DETER * 4);
  float*          dcay  = (float*)         alloc((size_t)N_DETER * 4);

  dim3 blk(256);

  prep_vec_kernel<<<dim3((N_DETER + 255) / 256), blk, 0, stream>>>(a_log, dt_p, dtv, dcay);
  pack_x1_kernel<<<dim3((unsigned)(((size_t)N_B * K1_PAD + 255) / 256)), blk, 0, stream>>>(stoch, action, x1);
  // Weights packed TRANSPOSED [N, Kpad] bf16 so GEMM B-tiles are K-contiguous.
  pack_wT_bf16<<<dim3((unsigned)(((size_t)N_HID * K1_PAD + 255) / 256)), blk, 0, stream>>>(
      W1, w1t, N_STOCH + N_ACT, K1_PAD, N_HID);
  pack_wT_bf16<<<dim3((unsigned)(((size_t)N_DETER * N_HID + 255) / 256)), blk, 0, stream>>>(
      W2, w2t, N_HID, N_HID, N_DETER);
  pack_wT_bf16<<<dim3((unsigned)(((size_t)N_DETER * 2 * N_DETER + 255) / 256)), blk, 0, stream>>>(
      Wg, wgt, 2 * N_DETER, 2 * N_DETER, N_DETER);
  pack_wT_bf16<<<dim3((unsigned)(((size_t)N_DETER * 2 * N_DETER + 255) / 256)), blk, 0, stream>>>(
      Wc, wct, 2 * N_DETER, 2 * N_DETER, N_DETER);

  // GEMM1: [B,1056] @ [1056,1024] + b1
  gemm_bf16_wmma<<<dim3(N_B / 128, N_HID / 256), blk, 0, stream>>>(
      x1, w1t, b1, hpre, N_B, N_HID, K1_PAD);
  act1_kernel<<<dim3(N_B), blk, 0, stream>>>(hpre, g1, xact);
  // GEMM2: [B,1024] @ [1024,2048] + b2
  gemm_bf16_wmma<<<dim3(N_B / 128, N_DETER / 256), blk, 0, stream>>>(
      xact, w2t, b2, inp, N_B, N_DETER, N_HID);
  lowrank_ssm_kernel<<<dim3(N_B), blk, 0, stream>>>(deter, Wu, Wv, dtv, dcay, inp, joint);
  // GEMM3/4: [B,4096] @ [4096,2048] + bias
  gemm_bf16_wmma<<<dim3(N_B / 128, N_DETER / 256), blk, 0, stream>>>(
      joint, wgt, bg, gpre, N_B, N_DETER, 2 * N_DETER);
  gemm_bf16_wmma<<<dim3(N_B / 128, N_DETER / 256), blk, 0, stream>>>(
      joint, wct, bc, cpre, N_B, N_DETER, 2 * N_DETER);
  final_kernel<<<dim3(N_B), blk, 0, stream>>>(gpre, cpre, deter, gn, out);
}
